// BasicBlock_export_3659312136801
// MI455X (gfx1250) — matrix operand order repaired and checked by reading, not compiled, not run on hardware
//
#include <hip/hip_runtime.h>

// ---------------------------------------------------------------------------
// Problem constants (from reference)
// ---------------------------------------------------------------------------
#define B_    16
#define T_    2048
#define DIN   1024
#define DPROJ 512
#define DOUT  1024
#define LORD  20
#define M_    (B_ * T_)   // 32768 rows for both GEMMs

typedef __attribute__((ext_vector_type(16))) __bf16 v16bf;
typedef __attribute__((ext_vector_type(8)))  float  v8f;

union Frag {
    uint4 q[2];
    v16bf v;
};

__device__ __forceinline__ unsigned short f2bf(float f) {
    unsigned int u = __float_as_uint(f);
    // round-to-nearest-even
    unsigned int r = u + 0x7FFFu + ((u >> 16) & 1u);
    return (unsigned short)(r >> 16);
}

__device__ __forceinline__ unsigned int pack2(float a, float b) {
    return (unsigned int)f2bf(a) | ((unsigned int)f2bf(b) << 16);
}

// Flat shared-memory pointer -> 32-bit LDS byte address (ISA: for LDS aperture
// addresses, the LDS address is the low 32 bits).
__device__ __forceinline__ unsigned lds_addr32(const void* p) {
    return (unsigned)(unsigned long long)(uintptr_t)p;
}

// CDNA5 async DMA: copy 16 bytes global -> LDS, tracked by ASYNCcnt.
// GVS mode: 64-bit uniform base in SGPRs + per-lane 32-bit byte offset.
__device__ __forceinline__ void async_ld_b128(unsigned ldsaddr, unsigned goff,
                                              const void* base) {
    asm volatile("global_load_async_to_lds_b128 %0, %1, %2"
                 :: "v"(ldsaddr), "v"(goff), "s"(base)
                 : "memory");
}

__device__ __forceinline__ void wait_async0() {
    asm volatile("s_wait_asynccnt 0" ::: "memory");
}

// ---------------------------------------------------------------------------
// Weight transpose + fp32 -> bf16:  src[K][N] row-major  ->  dst[N][K]
// ---------------------------------------------------------------------------
__global__ void transpose_to_bf16(const float* __restrict__ src,
                                  __bf16* __restrict__ dst, int K, int N) {
    int idx = blockIdx.x * blockDim.x + threadIdx.x;
    if (idx >= K * N) return;
    int k = idx / N;
    int n = idx % N;
    unsigned short b = f2bf(src[idx]);
    dst[(size_t)n * K + k] = *reinterpret_cast<__bf16*>(&b);
}

// ---------------------------------------------------------------------------
// WMMA GEMM: D[M x NOUT] = A[M x K] * Bt[NOUT x K]^T (+bias, ReLU)
//   Block tile 128(M) x 64(N), BK = 32, 256 threads = 8 waves (wave32).
//   Wave (wm,wn) in 4x2 arrangement computes a 32x32 region = 2x2 WMMA tiles.
//   Double-buffered LDS: async DMA for stage k+1 overlaps WMMA on stage k,
//   one barrier per stage.
//   A_F32: A is fp32 in global (VALU bf16 convert while staging),
//   else A is bf16 row-major in global (async DMA).  B always async DMA.
// ---------------------------------------------------------------------------
template <bool A_F32, bool FUSE_BIAS_RELU, int K, int NOUT>
__global__ __launch_bounds__(256) void wmma_gemm(
    const void* __restrict__ Aglob, const __bf16* __restrict__ Bt,
    const float* __restrict__ bias, float* __restrict__ Dglob) {
    const int tid     = threadIdx.x;
    const int lane    = tid & 31;
    const int wave    = tid >> 5;
    const int wm      = wave & 3;   // 0..3  -> M sub-tile of 32 rows
    const int wn      = wave >> 2;  // 0..1  -> N sub-tile of 32 cols
    const int lane_lo = lane & 15;
    const int hi      = lane >> 4;

    const int m0 = blockIdx.x * 128;
    const int n0 = blockIdx.y * 64;

    __shared__ __bf16 As[2][128][32];
    __shared__ __bf16 Bs[2][64][32];

    v8f c[2][2] = {};

    // staging roles
    const int ar  = tid >> 1;  // 0..127 : A row (fp32 convert path)
    const int akh = tid & 1;   // 0..1   : 16-wide K half
    const int br  = tid >> 2;  // 0..63  : B row (n)
    const int bch = tid & 3;   // 0..3   : 8-half (16B) chunk

    // Loop-invariant LDS byte addresses for async DMA destinations
    unsigned ldsB[2], ldsA0[2], ldsA1[2];
#pragma unroll
    for (int bufi = 0; bufi < 2; ++bufi) {
        ldsB[bufi]  = lds_addr32(&Bs[bufi][br][bch * 8]);
        ldsA0[bufi] = lds_addr32(&As[bufi][tid >> 2][(tid & 3) * 8]);
        ldsA1[bufi] = lds_addr32(&As[bufi][64 + (tid >> 2)][(tid & 3) * 8]);
    }

    auto stage = [&](int buf, int kk) {
        // ---- B tile (64 x 32 halfs): async DMA global -> LDS ----
        unsigned goff = (unsigned)((((size_t)(n0 + br) * K) + kk + bch * 8) *
                                   sizeof(__bf16));
        async_ld_b128(ldsB[buf], goff, (const void*)Bt);

        // ---- A tile (128 x 32) ----
        if (A_F32) {
            const float* A = (const float*)Aglob;
            const float4* src =
                (const float4*)(A + (size_t)(m0 + ar) * K + kk + akh * 16);
            float4 f0 = src[0], f1 = src[1], f2 = src[2], f3 = src[3];
            if (kk + 32 < K)  // prefetch next K-stage (global_prefetch_b8)
                __builtin_prefetch((const char*)src + 32 * sizeof(float), 0, 3);
            uint4 p0, p1;
            p0.x = pack2(f0.x, f0.y); p0.y = pack2(f0.z, f0.w);
            p0.z = pack2(f1.x, f1.y); p0.w = pack2(f1.z, f1.w);
            p1.x = pack2(f2.x, f2.y); p1.y = pack2(f2.z, f2.w);
            p1.z = pack2(f3.x, f3.y); p1.w = pack2(f3.z, f3.w);
            *(uint4*)&As[buf][ar][akh * 16]     = p0;
            *(uint4*)&As[buf][ar][akh * 16 + 8] = p1;
        } else {
            // bf16 A: two async 16B chunks per thread (512 chunks total)
            const int arow = tid >> 2;
            const int acc  = tid & 3;
            unsigned g0 = (unsigned)((((size_t)(m0 + arow) * K) + kk +
                                      acc * 8) * sizeof(__bf16));
            unsigned g1 = (unsigned)((((size_t)(m0 + 64 + arow) * K) + kk +
                                      acc * 8) * sizeof(__bf16));
            async_ld_b128(ldsA0[buf], g0, Aglob);
            async_ld_b128(ldsA1[buf], g1, Aglob);
        }
    };

    // ---- prologue: fill buffer 0 with K-stage 0 ----
    stage(0, 0);
    wait_async0();
    __syncthreads();

    int cur = 0;
    for (int k0 = 0; k0 < K; k0 += 32) {
        const int nxt = cur ^ 1;
        // issue DMA for the next stage; overlaps with WMMA below
        if (k0 + 32 < K) stage(nxt, k0 + 32);

        // ---- fragments from current buffer ----
        // A (16-bit 16x32): lane holds row = lane&15; halfs K[hi*8..+7] and
        // K[16+hi*8..+7]  (ISA 7.12.2 16-bit A layout)
        Frag a[2];
#pragma unroll
        for (int i = 0; i < 2; ++i) {
            int row = wm * 32 + i * 16 + lane_lo;
            a[i].q[0] = *(const uint4*)&As[cur][row][hi * 8];
            a[i].q[1] = *(const uint4*)&As[cur][row][16 + hi * 8];
        }
        // B (32x16): lane holds col = lane&15; halfs K[hi*8..+7] and K[16+hi*8..+7]
        Frag b[2];
#pragma unroll
        for (int j = 0; j < 2; ++j) {
            int col = wn * 32 + j * 16 + lane_lo;
            b[j].q[0] = *(const uint4*)&Bs[cur][col][hi * 8];
            b[j].q[1] = *(const uint4*)&Bs[cur][col][16 + hi * 8];
        }

#pragma unroll
        for (int i = 0; i < 2; ++i)
#pragma unroll
            for (int j = 0; j < 2; ++j)
                c[i][j] = __builtin_amdgcn_wmma_f32_16x16x32_bf16(
                    false, a[i].v, false, b[j].v, (short)0, c[i][j],
                    false, false);

        wait_async0();    // next-stage DMA writes to LDS complete
        __syncthreads();  // all waves done reading cur + see next buffer
        cur = nxt;
    }

    // ---- epilogue: C/D layout = VGPR r -> row r + 8*hi, col = lane&15 ----
    float* dp = Dglob + (size_t)(m0 + wm * 32 + hi * 8) * NOUT +
                (n0 + wn * 32 + lane_lo);
#pragma unroll
    for (int i = 0; i < 2; ++i) {
#pragma unroll
        for (int j = 0; j < 2; ++j) {
            float bv = 0.0f;
            if (FUSE_BIAS_RELU) bv = bias[n0 + wn * 32 + j * 16 + lane_lo];
#pragma unroll
            for (int rr = 0; rr < 8; ++rr) {
                float v = c[i][j][rr];
                if (FUSE_BIAS_RELU) v = fmaxf(v + bv, 0.0f);
                dp[(size_t)(i * 16 + rr) * NOUT + j * 16] = v;
            }
        }
    }
}

// ---------------------------------------------------------------------------
// Depthwise causal conv (L=20) + residual; emits m as bf16 row-major [m][p]
// One thread per (b, p); register sliding window over t.
// ---------------------------------------------------------------------------
__global__ __launch_bounds__(256) void conv_residual_kernel(
    const float* __restrict__ x, const float* __restrict__ cache,
    const float* __restrict__ cw, __bf16* __restrict__ mOut) {
    int gid = blockIdx.x * blockDim.x + threadIdx.x;  // 0..B*DPROJ-1
    int b = gid / DPROJ;
    int p = gid % DPROJ;

    float w[LORD];
#pragma unroll
    for (int l = 0; l < LORD; ++l) w[l] = cw[p * LORD + l];

    float win[LORD];
#pragma unroll
    for (int l = 0; l < LORD - 1; ++l)
        win[l] = cache[((size_t)b * DPROJ + p) * (LORD - 1) + l];

    const float* xp = x + (size_t)b * T_ * DPROJ + p;
    __bf16* mp = mOut + (size_t)b * T_ * DPROJ + p;

    for (int t = 0; t < T_; ++t) {
        float xv = xp[(size_t)t * DPROJ];
        win[LORD - 1] = xv;
        float acc = 0.0f;
#pragma unroll
        for (int l = 0; l < LORD; ++l) acc = fmaf(win[l], w[l], acc);
        float mv = acc + xv;  // residual
        unsigned short bv = f2bf(mv);
        mp[(size_t)t * DPROJ] = *reinterpret_cast<__bf16*>(&bv);
#pragma unroll
        for (int l = 0; l < LORD - 1; ++l) win[l] = win[l + 1];
    }
}

// ---------------------------------------------------------------------------
// out_cache[b][p][l] = x[b][T-19+l][p]
// ---------------------------------------------------------------------------
__global__ void out_cache_kernel(const float* __restrict__ x,
                                 float* __restrict__ oc) {
    int idx = blockIdx.x * blockDim.x + threadIdx.x;
    const int total = B_ * DPROJ * (LORD - 1);
    if (idx >= total) return;
    int l = idx % (LORD - 1);
    int p = (idx / (LORD - 1)) % DPROJ;
    int b = idx / (DPROJ * (LORD - 1));
    int t = T_ - (LORD - 1) + l;
    oc[idx] = x[((size_t)b * T_ + t) * DPROJ + p];
}

// ---------------------------------------------------------------------------
// Launch
// ---------------------------------------------------------------------------
extern "C" void kernel_launch(void* const* d_in, const int* in_sizes, int n_in,
                              void* d_out, int out_size, void* d_ws,
                              size_t ws_size, hipStream_t stream) {
    const float* input    = (const float*)d_in[0];  // [B][T][DIN]
    const float* in_cache = (const float*)d_in[1];  // [B][DPROJ][19]
    const float* W_lin    = (const float*)d_in[2];  // [DIN][DPROJ]
    const float* conv_w   = (const float*)d_in[3];  // [DPROJ][LORD]
    const float* W_aff    = (const float*)d_in[4];  // [DPROJ][DOUT]
    const float* b_aff    = (const float*)d_in[5];  // [DOUT]

    float* out       = (float*)d_out;                     // [B][T][DOUT]
    float* out_cache = out + (size_t)M_ * DOUT;           // [B][DPROJ][19]

    // workspace layout
    float*  x_ws = (float*)d_ws;                               // M_ x DPROJ f32
    __bf16* m_ws = (__bf16*)(x_ws + (size_t)M_ * DPROJ);       // M_ x DPROJ bf16
    __bf16* wlt  = m_ws + (size_t)M_ * DPROJ;                  // [DPROJ][DIN]
    __bf16* wat  = wlt + (size_t)DPROJ * DIN;                  // [DOUT][DPROJ]

    // 1. weight prep (transpose + bf16 convert)
    transpose_to_bf16<<<(DIN * DPROJ + 255) / 256, 256, 0, stream>>>(
        W_lin, wlt, DIN, DPROJ);
    transpose_to_bf16<<<(DPROJ * DOUT + 255) / 256, 256, 0, stream>>>(
        W_aff, wat, DPROJ, DOUT);

    // 2. GEMM1: x = input @ W_lin   (A fp32 -> bf16 on stage)
    wmma_gemm<true, false, DIN, DPROJ>
        <<<dim3(M_ / 128, DPROJ / 64), 256, 0, stream>>>(
            (const void*)input, wlt, nullptr, x_ws);

    // 3. depthwise conv + residual -> m (bf16)
    conv_residual_kernel<<<(B_ * DPROJ) / 256, 256, 0, stream>>>(
        x_ws, in_cache, conv_w, m_ws);

    // 4. GEMM2: out = relu(m @ W_aff + b_aff)
    wmma_gemm<false, true, DPROJ, DOUT>
        <<<dim3(M_ / 128, DOUT / 64), 256, 0, stream>>>(
            (const void*)m_ws, wat, b_aff, out);

    // 5. out_cache copy
    const int ctot = B_ * DPROJ * (LORD - 1);
    out_cache_kernel<<<(ctot + 255) / 256, 256, 0, stream>>>(x_ws, out_cache);
}
